// GATGraphRegression_36885179138613
// MI455X (gfx1250) — matrix operand order repaired and checked by reading, not compiled, not run on hardware
//
#include <hip/hip_runtime.h>
#include <hip/hip_bf16.h>

#define NN 50000
#define HH 4
#define CC 64
#define GG 256
#define DHC 256            // H*C
#define NEG_SLOPE 0.2f
#define LDSW 264           // padded LDS row stride (bf16 elems): 528B = 33*16 -> conflict-free

typedef __attribute__((ext_vector_type(16))) __bf16 v16bf;
typedef __attribute__((ext_vector_type(8)))  __bf16 v8bf;
typedef __attribute__((ext_vector_type(8)))  float  v8f;
typedef __attribute__((ext_vector_type(4)))  unsigned int v4u;
typedef __attribute__((ext_vector_type(8)))  int v8i_;
typedef __attribute__((ext_vector_type(4)))  int v4i_;

// ---------------- elementwise casts ----------------

__global__ void cast_bf16_kernel(const float* __restrict__ in, __bf16* __restrict__ out, int n) {
    int i = blockIdx.x * blockDim.x + threadIdx.x;
    if (i < n) out[i] = (__bf16)in[i];
}

// W[k, n] (row-major, 256 cols) -> Wt[n, k] bf16
__global__ void cast_wt_kernel(const float* __restrict__ W, __bf16* __restrict__ Wt, int K) {
    int i = blockIdx.x * blockDim.x + threadIdx.x;
    if (i >= K * DHC) return;
    int k = i / DHC, n = i % DHC;
    Wt[(size_t)n * K + k] = (__bf16)W[i];
}

// ---------------- WMMA GEMM: C[N,256] = A[N,K] @ W[K,256] ----------------
// grid = N/16 blocks, 512 threads (16 waves). Wave w computes col tile [16w,16w+16).
// A tile (16 x K bf16) is staged into LDS by the Tensor Data Mover with hardware
// row padding (row stride 528B) to keep the ds_load_b128 fragment reads conflict-free.

__global__ __launch_bounds__(512)
void gemm_wmma_kernel(const __bf16* __restrict__ A, const __bf16* __restrict__ Wt,
                      float* __restrict__ C, int K) {
    __shared__ __bf16 ldsA[16 * LDSW];
    const int row0 = blockIdx.x * 16;
    const int tid  = threadIdx.x;

    if (tid < 32) {   // wave 0 issues one TDM DMA for the whole block
        unsigned lds_base = (unsigned)(unsigned long long)(const void*)&ldsA[0];
        unsigned long long ga =
            (unsigned long long)(const void*)(A + (size_t)row0 * K);
        // D# group 0: count=1, lds_addr, global_addr[56:0], type=2 ("image")
        v4u g0;
        g0.x = 1u;
        g0.y = lds_base;
        g0.z = (unsigned)ga;
        g0.w = (unsigned)((ga >> 32) & 0x01FFFFFFu) | (2u << 30);
        // D# group 1: data_size=2B, pad after each row (K/2 dwords) by 4 dwords (16B)
        const int interval = (K == 256) ? 6 : 5;   // 2^(i+1) dwords = one row
        v8i_ g1;
        g1[0] = (1 << 16) | (1 << 20) | (interval << 22) | (3 << 25);
        g1[1] = (K & 0xFFFF) << 16;        // tensor_dim0[15:0]  (bits 63:48)
        g1[2] = (16 & 0xFFFF) << 16;       // tensor_dim1[15:0]  (bits 95:80)
        g1[3] = (K << 16);                 // tile_dim0          (bits 127:112)
        g1[4] = 16;                        // tile_dim1=16, tile_dim2=0
        g1[5] = K;                         // tensor_dim0_stride[31:0]
        g1[6] = 0;
        g1[7] = 0;
        v4i_ gz4 = {0, 0, 0, 0};
        v8i_ gz8 = {0, 0, 0, 0, 0, 0, 0, 0};
        __builtin_amdgcn_tensor_load_to_lds(g0, g1, gz4, gz4, gz8, 0);
        __builtin_amdgcn_s_wait_tensorcnt(0);
    }
    __syncthreads();

    const int wave = tid >> 5;
    const int lane = tid & 31;
    const int col0 = wave * 16;
    const int arow = lane & 15;          // A row / C column within tile
    const int col  = col0 + arow;
    const int kA   = (lane < 16) ? 0 : 16;  // A: hi lanes take K+16..31
    const int kB   = (lane < 16) ? 0 : 16;  // B: hi lanes take K+16..31

    v8f acc = {};
    for (int k0 = 0; k0 < K; k0 += 32) {
        union { v16bf v; v8bf h[2]; } au;
        const __bf16* ap = &ldsA[arow * LDSW + k0 + kA];
        au.h[0] = *reinterpret_cast<const v8bf*>(ap);        // K = kA+0..7
        au.h[1] = *reinterpret_cast<const v8bf*>(ap + 8);    // K = kA+8..15
        v16bf bfrag = *reinterpret_cast<const v16bf*>(Wt + (size_t)col * K + k0 + kB);
        acc = __builtin_amdgcn_wmma_f32_16x16x32_bf16(
            false, au.v, false, bfrag, (short)0, acc, false, false);
    }

    // C/D layout: VGPR v -> M = v (lanes 0-15) or v+8 (lanes 16-31), N = lane&15
#pragma unroll
    for (int v = 0; v < 8; ++v) {
        int r = (lane < 16) ? v : v + 8;
        C[(size_t)(row0 + r) * DHC + col] = acc[v];
    }
}

// ---------------- attention scalars ----------------

__global__ void alpha_kernel(const float* __restrict__ H, const float* __restrict__ a_src,
                             const float* __restrict__ a_dst,
                             float* __restrict__ as_, float* __restrict__ ad_) {
    int i = blockIdx.x * blockDim.x + threadIdx.x;   // over N*H
    if (i >= NN * HH) return;
    int n = i >> 2, hd = i & 3;
    const float* hp = H + (size_t)n * DHC + hd * CC;
    float s = 0.f, d = 0.f;
    for (int c = 0; c < CC; ++c) {
        float v = hp[c];
        s += v * a_src[hd * CC + c];
        d += v * a_dst[hd * CC + c];
    }
    as_[i] = s; ad_[i] = d;
}

__global__ void neginf_kernel(float* __restrict__ p, int n) {
    int i = blockIdx.x * blockDim.x + threadIdx.x;
    if (i < n) p[i] = -INFINITY;
}

__device__ inline void atomicMaxF(float* addr, float val) {
    if (val >= 0.f) atomicMax((int*)addr, __float_as_int(val));
    else            atomicMin((unsigned int*)addr, __float_as_uint(val));
}

__device__ inline void edge_pair(const int* __restrict__ ei, int e, int E, int& src, int& dst) {
    if (e < E) { src = ei[e]; dst = ei[E + e]; }
    else       { src = dst = e - E; }              // self loops
}

__global__ void edge_max_kernel(const int* __restrict__ ei, int E,
                                const float* __restrict__ as_, const float* __restrict__ ad_,
                                float* __restrict__ m) {
    int e = blockIdx.x * blockDim.x + threadIdx.x;
    if (e >= E + NN) return;
    int src, dst; edge_pair(ei, e, E, src, dst);
    float4 s = *reinterpret_cast<const float4*>(as_ + (size_t)src * HH);
    float4 d = *reinterpret_cast<const float4*>(ad_ + (size_t)dst * HH);
    float v[4] = {s.x + d.x, s.y + d.y, s.z + d.z, s.w + d.w};
#pragma unroll
    for (int h = 0; h < HH; ++h) {
        float x = v[h] > 0.f ? v[h] : NEG_SLOPE * v[h];
        atomicMaxF(&m[(size_t)dst * HH + h], x);
    }
}

__global__ void edge_den_kernel(const int* __restrict__ ei, int E,
                                const float* __restrict__ as_, const float* __restrict__ ad_,
                                const float* __restrict__ m, float* __restrict__ den) {
    int e = blockIdx.x * blockDim.x + threadIdx.x;
    if (e >= E + NN) return;
    int src, dst; edge_pair(ei, e, E, src, dst);
    float4 s  = *reinterpret_cast<const float4*>(as_ + (size_t)src * HH);
    float4 d  = *reinterpret_cast<const float4*>(ad_ + (size_t)dst * HH);
    float4 mm = *reinterpret_cast<const float4*>(m   + (size_t)dst * HH);
    float v[4] = {s.x + d.x, s.y + d.y, s.z + d.z, s.w + d.w};
    float mv[4] = {mm.x, mm.y, mm.z, mm.w};
#pragma unroll
    for (int h = 0; h < HH; ++h) {
        float x = v[h] > 0.f ? v[h] : NEG_SLOPE * v[h];
        atomicAdd(&den[(size_t)dst * HH + h], __expf(x - mv[h]));
    }
}

// one wave per edge: gather h[src], scale by per-head alpha, atomic-add into AGG[dst]
__global__ __launch_bounds__(256)
void edge_aggr_kernel(const int* __restrict__ ei, int E,
                      const float* __restrict__ as_, const float* __restrict__ ad_,
                      const float* __restrict__ m, const float* __restrict__ den,
                      const float* __restrict__ H, float* __restrict__ AGG) {
    int gw = (blockIdx.x * blockDim.x + threadIdx.x) >> 5;   // wave-uniform
    int lane = threadIdx.x & 31;
    if (gw >= E + NN) return;
    int src, dst; edge_pair(ei, gw, E, src, dst);
    float4 s  = *reinterpret_cast<const float4*>(as_ + (size_t)src * HH);
    float4 d  = *reinterpret_cast<const float4*>(ad_ + (size_t)dst * HH);
    float4 mm = *reinterpret_cast<const float4*>(m   + (size_t)dst * HH);
    float4 dd = *reinterpret_cast<const float4*>(den + (size_t)dst * HH);
    float v[4]  = {s.x + d.x, s.y + d.y, s.z + d.z, s.w + d.w};
    float mv[4] = {mm.x, mm.y, mm.z, mm.w};
    float dv[4] = {dd.x, dd.y, dd.z, dd.w};
    float w[4];
#pragma unroll
    for (int h = 0; h < HH; ++h) {
        float x = v[h] > 0.f ? v[h] : NEG_SLOPE * v[h];
        w[h] = __expf(x - mv[h]) / dv[h];
    }
    const float* hs = H + (size_t)src * DHC;
    float* op = AGG + (size_t)dst * DHC;
#pragma unroll
    for (int j = 0; j < 8; ++j) {
        int c = j * 32 + lane;                    // coalesced across lanes
        atomicAdd(&op[c], hs[c] * w[c >> 6]);
    }
}

__global__ void finalize_kernel(float* __restrict__ AGG, const float* __restrict__ b) {
    int i = blockIdx.x * blockDim.x + threadIdx.x;
    if (i >= NN * DHC) return;
    float x = AGG[i] + b[i & (DHC - 1)];
    AGG[i] = x > 0.f ? x : (__expf(x) - 1.f);     // ELU
}

// ---------------- pooling + head ----------------

__global__ void pool_kernel(const float* __restrict__ AGG, const int* __restrict__ batch,
                            float* __restrict__ pooled) {
    int i = blockIdx.x * blockDim.x + threadIdx.x;
    if (i >= NN * DHC) return;
    int n = i >> 8, c = i & (DHC - 1);
    atomicAdd(&pooled[(size_t)batch[n] * DHC + c], AGG[i]);
}

__global__ void cnt_kernel(const int* __restrict__ batch, float* __restrict__ cnt) {
    int i = blockIdx.x * blockDim.x + threadIdx.x;
    if (i < NN) atomicAdd(&cnt[batch[i]], 1.0f);
}

__global__ __launch_bounds__(256)
void head_kernel(const float* __restrict__ pooled, const float* __restrict__ cnt,
                 const float* __restrict__ l1w, const float* __restrict__ l1b,
                 const float* __restrict__ l2w, const float* __restrict__ l2b,
                 float* __restrict__ out) {
    int g = threadIdx.x;
    if (g >= GG) return;
    float inv = 1.0f / fmaxf(cnt[g], 1.0f);
    float hid[CC];
#pragma unroll
    for (int j = 0; j < CC; ++j) hid[j] = l1b[j];
    for (int c = 0; c < DHC; ++c) {
        float pc = pooled[(size_t)g * DHC + c] * inv;
#pragma unroll
        for (int j = 0; j < CC; ++j) hid[j] += pc * l1w[c * CC + j];
    }
    float acc = l2b[0];
#pragma unroll
    for (int j = 0; j < CC; ++j) acc += fmaxf(hid[j], 0.0f) * l2w[j];
    out[g] = acc;
}

// ---------------- driver ----------------

extern "C" void kernel_launch(void* const* d_in, const int* in_sizes, int n_in,
                              void* d_out, int out_size, void* d_ws, size_t ws_size,
                              hipStream_t stream) {
    const float* x     = (const float*)d_in[0];
    const int*   ei    = (const int*)d_in[1];
    const int*   batch = (const int*)d_in[2];
    const float* W[3]   = {(const float*)d_in[3],  (const float*)d_in[7],  (const float*)d_in[11]};
    const float* asw[3] = {(const float*)d_in[4],  (const float*)d_in[8],  (const float*)d_in[12]};
    const float* adw[3] = {(const float*)d_in[5],  (const float*)d_in[9],  (const float*)d_in[13]};
    const float* bb[3]  = {(const float*)d_in[6],  (const float*)d_in[10], (const float*)d_in[14]};
    const float* l1w = (const float*)d_in[15];
    const float* l1b = (const float*)d_in[16];
    const float* l2w = (const float*)d_in[17];
    const float* l2b = (const float*)d_in[18];

    const int E  = in_sizes[1] / 2;
    const int EP = E + NN;

    char* base = (char*)d_ws;
    size_t off = 0;
    auto alloc = [&](size_t bytes) -> void* {
        off = (off + 255) & ~(size_t)255;
        void* p = base + off;
        off += bytes;
        return p;
    };
    __bf16* A_bf   = (__bf16*)alloc((size_t)NN * DHC * 2);
    __bf16* Wt_bf  = (__bf16*)alloc((size_t)DHC * DHC * 2);
    float*  Hbuf   = (float*)alloc((size_t)NN * DHC * 4);
    float*  AGG    = (float*)alloc((size_t)NN * DHC * 4);
    float*  as_    = (float*)alloc((size_t)NN * HH * 4);
    float*  ad_    = (float*)alloc((size_t)NN * HH * 4);
    float*  mbuf   = (float*)alloc((size_t)NN * HH * 4);
    float*  den    = (float*)alloc((size_t)NN * HH * 4);
    float*  pooled = (float*)alloc((size_t)GG * DHC * 4);
    float*  cnt    = (float*)alloc((size_t)GG * 4);

    for (int l = 0; l < 3; ++l) {
        const int K = (l == 0) ? 128 : 256;
        const float* inx = (l == 0) ? x : AGG;
        const int elems = NN * K;

        cast_bf16_kernel<<<(elems + 255) / 256, 256, 0, stream>>>(inx, A_bf, elems);
        cast_wt_kernel<<<(K * DHC + 255) / 256, 256, 0, stream>>>(W[l], Wt_bf, K);
        gemm_wmma_kernel<<<NN / 16, 512, 0, stream>>>(A_bf, Wt_bf, Hbuf, K);
        alpha_kernel<<<(NN * HH + 255) / 256, 256, 0, stream>>>(Hbuf, asw[l], adw[l], as_, ad_);

        neginf_kernel<<<(NN * HH + 255) / 256, 256, 0, stream>>>(mbuf, NN * HH);
        (void)hipMemsetAsync(den, 0, (size_t)NN * HH * 4, stream);
        (void)hipMemsetAsync(AGG, 0, (size_t)NN * DHC * 4, stream);  // after cast_bf16 consumed AGG

        edge_max_kernel<<<(EP + 255) / 256, 256, 0, stream>>>(ei, E, as_, ad_, mbuf);
        edge_den_kernel<<<(EP + 255) / 256, 256, 0, stream>>>(ei, E, as_, ad_, mbuf, den);
        edge_aggr_kernel<<<(EP + 7) / 8, 256, 0, stream>>>(ei, E, as_, ad_, mbuf, den, Hbuf, AGG);
        finalize_kernel<<<(NN * DHC + 255) / 256, 256, 0, stream>>>(AGG, bb[l]);
    }

    (void)hipMemsetAsync(pooled, 0, (size_t)GG * DHC * 4, stream);
    (void)hipMemsetAsync(cnt, 0, (size_t)GG * 4, stream);
    pool_kernel<<<(NN * DHC + 255) / 256, 256, 0, stream>>>(AGG, batch, pooled);
    cnt_kernel<<<(NN + 255) / 256, 256, 0, stream>>>(batch, cnt);
    head_kernel<<<1, 256, 0, stream>>>(pooled, cnt, l1w, l1b, l2w, l2b, (float*)d_out);
}
